// BiMamba_block_23407571763543
// MI455X (gfx1250) — matrix operand order repaired and checked by reading, not compiled, not run on hardware
//
#include <hip/hip_runtime.h>

typedef unsigned short u16;
typedef unsigned int   u32;
typedef __attribute__((ext_vector_type(16))) __bf16 v16bf;
typedef __attribute__((ext_vector_type(8)))  float  v8f;

#define B_SZ    8
#define L_SEQ   2048
#define D_MODEL 128
#define D_INNER 256
#define D_STATE 48
#define DT_RANK 8
#define NTOK    (B_SZ * L_SEQ)          // 16384
#define XP_N    112                      // 104 padded to 112 (mult of 16)

union FragAB { v16bf v; uint4 q[2]; };

__device__ __forceinline__ u16 f2bf(float f) {
    u32 u = __float_as_uint(f);
    u32 r = (u + 0x7FFFu + ((u >> 16) & 1u)) >> 16;
    return (u16)r;
}
__device__ __forceinline__ float siluf(float x) {
    return x / (1.0f + __expf(-x));
}

// ---------------- LayerNorm -> bf16 (one wave per token) ----------------
__global__ void ln_kernel(u16* __restrict__ xn, const float* __restrict__ x,
                          const float* __restrict__ w, const float* __restrict__ b) {
    int t    = blockIdx.x * 8 + (threadIdx.x >> 5);
    int lane = threadIdx.x & 31;
    float4 v  = ((const float4*)(x + (size_t)t * D_MODEL))[lane];
    float4 w4 = ((const float4*)w)[lane];
    float4 b4 = ((const float4*)b)[lane];
    float s  = v.x + v.y + v.z + v.w;
    float sq = v.x*v.x + v.y*v.y + v.z*v.z + v.w*v.w;
    #pragma unroll
    for (int m = 1; m < 32; m <<= 1) { s += __shfl_xor(s, m); sq += __shfl_xor(sq, m); }
    float mu   = s * (1.0f / D_MODEL);
    float var  = sq * (1.0f / D_MODEL) - mu * mu;
    float rstd = rsqrtf(var + 1e-5f);
    ushort4 o;
    o.x = f2bf((v.x - mu) * rstd * w4.x + b4.x);
    o.y = f2bf((v.y - mu) * rstd * w4.y + b4.y);
    o.z = f2bf((v.z - mu) * rstd * w4.z + b4.z);
    o.w = f2bf((v.w - mu) * rstd * w4.w + b4.w);
    ((ushort4*)(xn + (size_t)t * D_MODEL))[lane] = o;
}

// ---------------- f32 -> bf16 pack with row padding ----------------
__global__ void pack_bf16(u16* __restrict__ dst, const float* __restrict__ src,
                          int rows, int cols, int padRows) {
    int i = blockIdx.x * 256 + threadIdx.x;
    if (i >= padRows * cols) return;
    int r = i / cols;
    dst[i] = (r < rows) ? f2bf(src[i]) : (u16)0;
}

// ---------------- WMMA GEMM: C[MxN] = A[MxK](bf16) * W[NxK]^T(bf16) (+resid) ----
__global__ void wmma_gemm_bf16(float* __restrict__ C, const u16* __restrict__ A,
                               const u16* __restrict__ W, int M, int N, int K,
                               const float* __restrict__ resid) {
    int wave = threadIdx.x >> 5;
    int lane = threadIdx.x & 31;
    int hf   = lane >> 4;       // which 16-lane half
    int lsub = lane & 15;
    int tilesN = N >> 4;
    int tile = blockIdx.x * 4 + wave;
    if (tile >= (M >> 4) * tilesN) return;
    int m0 = (tile / tilesN) << 4;
    int n0 = (tile % tilesN) << 4;

    v8f acc = {};
    for (int k0 = 0; k0 < K; k0 += 32) {
        FragAB a, b;
        // A frag: lane half 0 -> K {0..7,16..23}; half 1 -> K {8..15,24..31}
        const uint4* ap = (const uint4*)(A + (size_t)(m0 + lsub) * K + k0 + hf * 8);
        a.q[0] = ap[0];
        a.q[1] = ap[2];
        // B frag: col n0+lsub; lane half 0 -> K {0..7,16..23}; half 1 -> K {8..15,24..31}
        const uint4* bp = (const uint4*)(W + (size_t)(n0 + lsub) * K + k0 + hf * 8);
        b.q[0] = bp[0];
        b.q[1] = bp[2];
        acc = __builtin_amdgcn_wmma_f32_16x16x32_bf16(false, a.v, false, b.v,
                                                      (short)0, acc, false, false);
    }
    #pragma unroll
    for (int j = 0; j < 8; ++j) {
        size_t idx = (size_t)(m0 + hf * 8 + j) * N + n0 + lsub;
        float v = acc[j];
        if (resid) v += resid[idx];
        C[idx] = v;
    }
}

// ---------------- depthwise causal conv (k=4) + SiLU, direction-aware ----------
__global__ void conv_silu_kernel(float* __restrict__ u, u16* __restrict__ ubf,
                                 const float* __restrict__ xz,
                                 const float* __restrict__ cw,
                                 const float* __restrict__ cb, int reverse) {
    int g = blockIdx.x * 256 + threadIdx.x;       // over NTOK * D_INNER
    int d = g & (D_INNER - 1);
    int t = g >> 8;                               // scan-order token = b*L + l
    int l = t & (L_SEQ - 1);
    int b = t >> 11;
    float acc = cb[d];
    #pragma unroll
    for (int k = 0; k < 4; ++k) {
        int ls = l - 3 + k;
        if (ls >= 0) {
            int p = reverse ? (L_SEQ - 1 - ls) : ls;
            acc = fmaf(cw[d * 4 + k], xz[((size_t)b * L_SEQ + p) * 512 + d], acc);
        }
    }
    float s = siluf(acc);
    u[g]   = s;
    ubf[g] = f2bf(s);
}

// ---------------- dt = softplus(dt_raw @ dt_w^T + dt_b) ----------------
__global__ void dt_kernel(float* __restrict__ dt, const float* __restrict__ x_dbl,
                          const float* __restrict__ dtw, const float* __restrict__ dtb) {
    int t = blockIdx.x;
    int d = threadIdx.x;
    const float* r = x_dbl + (size_t)t * XP_N;
    float acc = dtb[d];
    #pragma unroll
    for (int k = 0; k < DT_RANK; ++k) acc = fmaf(r[k], dtw[d * DT_RANK + k], acc);
    float sp = (acc > 20.f) ? acc : log1pf(__expf(acc));
    dt[(size_t)t * D_INNER + d] = sp;
}

// ---------------- selective scan: 16 lanes per (b,d), 3 states/lane ------------
__global__ void scan_kernel(float* __restrict__ ysum,
                            const float* __restrict__ x_dbl,
                            const float* __restrict__ dtbuf,
                            const float* __restrict__ ubuf,
                            const float* __restrict__ xz,
                            const float* __restrict__ A_log,
                            const float* __restrict__ Dp,
                            int reverse, int accumulate) {
    int b    = blockIdx.y;
    int d0   = blockIdx.x * 16;
    int tid  = threadIdx.x;
    int dgrp = tid >> 4;          // 0..15 -> which d
    int sgrp = tid & 15;          // 0..15 -> state group
    int d    = d0 + dgrp;
    int s0   = sgrp * 3;

    float An[3], h[3] = {0.f, 0.f, 0.f};
    #pragma unroll
    for (int i = 0; i < 3; ++i) An[i] = -__expf(A_log[d * D_STATE + s0 + i]);
    float Dpd = Dp[d];

    __shared__ float sBC[8][96];
    __shared__ float sdt[8][16];
    __shared__ float su[8][16];
    const size_t tb = (size_t)b * L_SEQ;

    for (int l0 = 0; l0 < L_SEQ; l0 += 8) {
        __syncthreads();
        for (int idx = tid; idx < 8 * 96; idx += 256) {
            int j = idx / 96, s = idx % 96;
            sBC[j][s] = x_dbl[(tb + l0 + j) * XP_N + 8 + s];   // B then C
        }
        for (int idx = tid; idx < 8 * 16; idx += 256) {
            int j = idx >> 4, dd = idx & 15;
            sdt[j][dd] = dtbuf[(tb + l0 + j) * D_INNER + d0 + dd];
            su[j][dd]  = ubuf [(tb + l0 + j) * D_INNER + d0 + dd];
        }
        __syncthreads();
        #pragma unroll 1
        for (int j = 0; j < 8; ++j) {
            int l = l0 + j;
            float dt  = sdt[j][dgrp];
            float uu  = su[j][dgrp];
            float dtu = dt * uu;
            float yp  = 0.f;
            #pragma unroll
            for (int i = 0; i < 3; ++i) {
                float dA = __expf(dt * An[i]);
                h[i] = dA * h[i] + dtu * sBC[j][s0 + i];
                yp = fmaf(h[i], sBC[j][48 + s0 + i], yp);
            }
            yp += __shfl_xor(yp, 1);
            yp += __shfl_xor(yp, 2);
            yp += __shfl_xor(yp, 4);
            yp += __shfl_xor(yp, 8);
            if (sgrp == 0) {
                int p = reverse ? (L_SEQ - 1 - l) : l;
                float z = xz[(tb + p) * 512 + D_INNER + d];
                float y = (yp + uu * Dpd) * siluf(z);
                size_t o = (tb + p) * D_INNER + d;
                ysum[o] = accumulate ? (ysum[o] + y) : y;
            }
        }
    }
}

// ---------------- host side ----------------
extern "C" void kernel_launch(void* const* d_in, const int* in_sizes, int n_in,
                              void* d_out, int out_size, void* d_ws, size_t ws_size,
                              hipStream_t stream) {
    const float* x        = (const float*)d_in[0];
    const float* ln_w     = (const float*)d_in[1];
    const float* ln_b     = (const float*)d_in[2];
    const float* in_proj  = (const float*)d_in[3];
    const float* out_proj = (const float*)d_in[4];
    // per-branch params: fwd at idx 5..11, bwd at idx 12..18
    const float* conv_w[2] = {(const float*)d_in[5],  (const float*)d_in[12]};
    const float* conv_b[2] = {(const float*)d_in[6],  (const float*)d_in[13]};
    const float* xp_w[2]   = {(const float*)d_in[7],  (const float*)d_in[14]};
    const float* dt_w[2]   = {(const float*)d_in[8],  (const float*)d_in[15]};
    const float* dt_b[2]   = {(const float*)d_in[9],  (const float*)d_in[16]};
    const float* A_log[2]  = {(const float*)d_in[10], (const float*)d_in[17]};
    const float* Dp[2]     = {(const float*)d_in[11], (const float*)d_in[18]};
    float* out = (float*)d_out;

    char* base = (char*)d_ws;
    size_t off = 0;
    auto alloc = [&](size_t bytes) -> char* {
        char* p = base + off;
        off += (bytes + 255) & ~(size_t)255;
        return p;
    };
    u16*   xn_bf  = (u16*)  alloc((size_t)NTOK * D_MODEL * 2);
    u16*   w_in   = (u16*)  alloc((size_t)512 * 128 * 2);
    u16*   w_xp   = (u16*)  alloc((size_t)XP_N * 256 * 2);
    u16*   w_out  = (u16*)  alloc((size_t)128 * 256 * 2);
    float* xz     = (float*)alloc((size_t)NTOK * 512 * 4);
    float* ubuf   = (float*)alloc((size_t)NTOK * D_INNER * 4);
    u16*   ubf    = (u16*)  alloc((size_t)NTOK * D_INNER * 2);
    float* x_dbl  = (float*)alloc((size_t)NTOK * XP_N * 4);
    float* dtbuf  = (float*)alloc((size_t)NTOK * D_INNER * 4);
    float* ysum   = (float*)alloc((size_t)NTOK * D_INNER * 4);
    u16*   ybf    = (u16*)  alloc((size_t)NTOK * D_INNER * 2);

    // 1. LayerNorm + cast
    ln_kernel<<<NTOK / 8, 256, 0, stream>>>(xn_bf, x, ln_w, ln_b);

    // 2. pack static weights
    pack_bf16<<<(512 * 128 + 255) / 256, 256, 0, stream>>>(w_in, in_proj, 512, 128, 512);
    pack_bf16<<<(128 * 256 + 255) / 256, 256, 0, stream>>>(w_out, out_proj, 128, 256, 128);

    // 3. in_proj GEMM: xz[NTOK x 512]
    {
        int tiles = (NTOK / 16) * (512 / 16);
        wmma_gemm_bf16<<<(tiles + 3) / 4, 128, 0, stream>>>(xz, xn_bf, w_in,
                                                            NTOK, 512, 128, nullptr);
    }

    // 4. two directional branches
    for (int br = 0; br < 2; ++br) {
        int rev = br;
        pack_bf16<<<(XP_N * 256 + 255) / 256, 256, 0, stream>>>(w_xp, xp_w[br], 104, 256, XP_N);
        conv_silu_kernel<<<NTOK * D_INNER / 256, 256, 0, stream>>>(ubuf, ubf, xz,
                                                                   conv_w[br], conv_b[br], rev);
        int tiles = (NTOK / 16) * (XP_N / 16);
        wmma_gemm_bf16<<<(tiles + 3) / 4, 128, 0, stream>>>(x_dbl, ubf, w_xp,
                                                            NTOK, XP_N, 256, nullptr);
        dt_kernel<<<NTOK, D_INNER, 0, stream>>>(dtbuf, x_dbl, dt_w[br], dt_b[br]);
        scan_kernel<<<dim3(D_INNER / 16, B_SZ), 256, 0, stream>>>(ysum, x_dbl, dtbuf, ubuf,
                                                                  xz, A_log[br], Dp[br],
                                                                  rev, /*accumulate=*/br);
    }

    // 5. cast combined y to bf16, out_proj GEMM with fused residual
    pack_bf16<<<(NTOK * D_INNER + 255) / 256, 256, 0, stream>>>(ybf, ysum,
                                                                NTOK, D_INNER, NTOK);
    {
        int tiles = (NTOK / 16) * (128 / 16);
        wmma_gemm_bf16<<<(tiles + 3) / 4, 128, 0, stream>>>(out, ybf, w_out,
                                                            NTOK, 128, 256, x);
    }
}
